// ASTGCNBlock_62062277427615
// MI455X (gfx1250) — matrix operand order repaired and checked by reading, not compiled, not run on hardware
//
#include <hip/hip_runtime.h>
#include <hip/hip_bf16.h>
#include <math.h>

typedef __bf16 bf16_t;
typedef __attribute__((ext_vector_type(16))) __bf16 v16bf;
typedef __attribute__((ext_vector_type(8)))  float  v8f;
typedef __attribute__((ext_vector_type(4)))  float  f32x4;

#define B_   16
#define T_   12
#define N_   2048
#define FIN  32
#define FOUT 64
#define KCH  3
#define C_   (T_*FIN)   /* 384 */

union ABf16 {
  uint4        q[2];
  unsigned int d[8];
  v16bf        v;
};
union Acc8 {
  v8f    v;
  f32x4  f4[2];
  float  f[8];
};

__device__ __forceinline__ unsigned int pk_mul_bf16(unsigned int a, unsigned int b) {
  unsigned int r;
  asm volatile("v_pk_mul_bf16 %0, %1, %2" : "=v"(r) : "v"(a), "v"(b));
  return r;
}

/* ---------------- generic f32 -> bf16 convert ---------------- */
__global__ void k_cvt_bf16(const float* __restrict__ src, bf16_t* __restrict__ dst, int n) {
  int i = blockIdx.x * blockDim.x + threadIdx.x;
  if (i < n) dst[i] = (bf16_t)src[i];
}

/* ---- tconv_w (O,CI,3,1) -> Wt[o][dt*64+ci] bf16 ---- */
__global__ void k_prep_wt(const float* __restrict__ w, bf16_t* __restrict__ wt) {
  int i = blockIdx.x * blockDim.x + threadIdx.x;
  if (i >= FOUT * 3 * FOUT) return;
  int o  = i / 192;
  int kk = i - o * 192;
  int dt = kk >> 6;
  int ci = kk & 63;
  wt[i] = (bf16_t)w[(o * FOUT + ci) * 3 + dt];
}

/* ---------------- qt/kt : (B,T,N) dots over F_IN ---------------- */
__global__ void k_qtkt(const float* __restrict__ x, const float* __restrict__ wq,
                       const float* __restrict__ wk, float* __restrict__ qt,
                       float* __restrict__ kt) {
  int idx = blockIdx.x * blockDim.x + threadIdx.x;   /* over B*T*N */
  const float* xr = x + (size_t)idx * FIN;
  float aq = 0.f, ak = 0.f;
#pragma unroll
  for (int f = 0; f < FIN; ++f) { aq += xr[f] * wq[f]; ak += xr[f] * wk[f]; }
  qt[idx] = aq;
  kt[idx] = ak;
}

/* ---------------- temporal attention softmax: ta (B,12,12) ---------------- */
__global__ void k_ta(const float* __restrict__ qt, const float* __restrict__ kt,
                     float* __restrict__ ta) {
  __shared__ float lg[T_ * T_];
  const int b = blockIdx.x;
  const int tid = threadIdx.x;
  const float scale = 0.02209708691207961f;  /* 1/sqrt(2048) */
  if (tid < T_ * T_) {
    int t = tid / T_, s = tid - t * T_;
    const float* q = qt + ((size_t)b * T_ + t) * N_;
    const float* k = kt + ((size_t)b * T_ + s) * N_;
    float acc = 0.f;
    for (int n = 0; n < N_; ++n) acc += q[n] * k[n];
    lg[tid] = acc * scale;
  }
  __syncthreads();
  if (tid < T_) {
    float m = -3.4e38f;
#pragma unroll
    for (int s = 0; s < T_; ++s) m = fmaxf(m, lg[tid * T_ + s]);
    float sum = 0.f;
#pragma unroll
    for (int s = 0; s < T_; ++s) sum += __expf(lg[tid * T_ + s] - m);
    float inv = 1.f / sum;
#pragma unroll
    for (int s = 0; s < T_; ++s)
      ta[(size_t)b * T_ * T_ + tid * T_ + s] = __expf(lg[tid * T_ + s] - m) * inv;
  }
}

/* ------- x_ta producer: writes xtaT bf16 [b][c][n], qsT/ksT [b][n][t] ------- */
__global__ void k_xta(const float* __restrict__ x, const float* __restrict__ ta,
                      const float* __restrict__ wq_s, const float* __restrict__ wk_s,
                      bf16_t* __restrict__ xtaT, float* __restrict__ qsT,
                      float* __restrict__ ksT) {
  __shared__ float taL[T_ * T_];
  __shared__ float wqL[FIN], wkL[FIN];
  __shared__ float tile[T_ * 8 * FIN];      /* [t][nl][f] */
  const int b  = blockIdx.y;
  const int n0 = blockIdx.x * 8;
  const int tid = threadIdx.x;
  if (tid < T_ * T_) taL[tid] = ta[(size_t)b * T_ * T_ + tid];
  else if (tid < T_ * T_ + FIN) wqL[tid - T_ * T_] = wq_s[tid - T_ * T_];
  else if (tid < T_ * T_ + 2 * FIN) wkL[tid - T_ * T_ - FIN] = wk_s[tid - T_ * T_ - FIN];
  __syncthreads();

  const int nl = tid >> 5;     /* 0..7 */
  const int f  = tid & 31;
  float xs[T_];
#pragma unroll
  for (int s = 0; s < T_; ++s)
    xs[s] = x[(((size_t)b * T_ + s) * N_ + n0 + nl) * FIN + f];
#pragma unroll
  for (int t = 0; t < T_; ++t) {
    float v = 0.f;
#pragma unroll
    for (int s = 0; s < T_; ++s) v += taL[t * T_ + s] * xs[s];
    tile[(t * 8 + nl) * FIN + f] = v;
  }
  __syncthreads();
  /* transposed bf16 write: xtaT[b][c][n] */
  for (int e = tid; e < C_ * 8; e += 256) {
    int c = e >> 3, n2 = e & 7;
    int t = c >> 5, f2 = c & 31;
    xtaT[((size_t)b * C_ + c) * N_ + n0 + n2] = (bf16_t)tile[(t * 8 + n2) * FIN + f2];
  }
  /* qs/ks */
  if (tid < T_ * 8) {
    int t = tid >> 3, n2 = tid & 7;
    float aq = 0.f, ak = 0.f;
#pragma unroll
    for (int ff = 0; ff < FIN; ++ff) {
      float v = tile[(t * 8 + n2) * FIN + ff];
      aq += v * wqL[ff];
      ak += v * wkL[ff];
    }
    qsT[((size_t)b * N_ + n0 + n2) * T_ + t] = aq;
    ksT[((size_t)b * N_ + n0 + n2) * T_ + t] = ak;
  }
}

/* ------- spatial softmax: P[b][i][j] bf16, 8 rows per 256-thread block ------- */
__global__ void k_sa(const float* __restrict__ qsT, const float* __restrict__ ksT,
                     bf16_t* __restrict__ P) {
  const int b = blockIdx.y;
  const int r = threadIdx.x >> 5;
  const int lane = threadIdx.x & 31;
  const int i = blockIdx.x * 8 + r;
  const float scale = 0.2886751345948129f;  /* 1/sqrt(12) */
  float q[T_];
#pragma unroll
  for (int t = 0; t < T_; ++t) q[t] = qsT[((size_t)b * N_ + i) * T_ + t];

  float lmax = -3.4e38f;
  for (int j = lane; j < N_; j += 32) {
    const float* kr = ksT + ((size_t)b * N_ + j) * T_;
    float l = 0.f;
#pragma unroll
    for (int t = 0; t < T_; ++t) l += q[t] * kr[t];
    lmax = fmaxf(lmax, l * scale);
  }
#pragma unroll
  for (int m = 1; m < 32; m <<= 1) lmax = fmaxf(lmax, __shfl_xor(lmax, m, 32));

  float lsum = 0.f;
  for (int j = lane; j < N_; j += 32) {
    const float* kr = ksT + ((size_t)b * N_ + j) * T_;
    float l = 0.f;
#pragma unroll
    for (int t = 0; t < T_; ++t) l += q[t] * kr[t];
    lsum += __expf(l * scale - lmax);
  }
#pragma unroll
  for (int m = 1; m < 32; m <<= 1) lsum += __shfl_xor(lsum, m, 32);
  float inv = 1.f / lsum;

  for (int j = lane; j < N_; j += 32) {
    const float* kr = ksT + ((size_t)b * N_ + j) * T_;
    float l = 0.f;
#pragma unroll
    for (int t = 0; t < T_; ++t) l += q[t] * kr[t];
    P[((size_t)b * N_ + i) * N_ + j] = (bf16_t)(__expf(l * scale - lmax) * inv);
  }
}

/* ------------- main Chebyshev GEMM: rhs[k][b][c][i] (WMMA bf16) -------------
   One wave computes a 16-row i-tile x TWO 16-col c-tiles, 3 cheb orders
   -> 6 accumulators, 6 WMMAs per K-step, A fragments built once via
   v_pk_mul_bf16 and reused.  Double-buffered tile registers so global
   loads for step j+1 overlap the pk_mul/WMMA work of step j.
   __launch_bounds__(32,1): one wave per SIMD -> full VGPR budget, no spills. */
__device__ __forceinline__ void gemm_load(const bf16_t* prow, const bf16_t* crow0,
                                          const bf16_t* crow1, const bf16_t* crow2,
                                          const bf16_t* brow0, const bf16_t* brow1,
                                          int j0, ABf16& pa, ABf16& c0v, ABf16& c1v,
                                          ABf16& c2v, ABf16& b0, ABf16& b1) {
  pa.q[0]  = *(const uint4*)(prow + j0);   pa.q[1]  = *(const uint4*)(prow + j0 + 16);
  c0v.q[0] = *(const uint4*)(crow0 + j0);  c0v.q[1] = *(const uint4*)(crow0 + j0 + 16);
  c1v.q[0] = *(const uint4*)(crow1 + j0);  c1v.q[1] = *(const uint4*)(crow1 + j0 + 16);
  c2v.q[0] = *(const uint4*)(crow2 + j0);  c2v.q[1] = *(const uint4*)(crow2 + j0 + 16);
  b0.q[0]  = *(const uint4*)(brow0 + j0);  b0.q[1]  = *(const uint4*)(brow0 + j0 + 16);
  b1.q[0]  = *(const uint4*)(brow1 + j0);  b1.q[1]  = *(const uint4*)(brow1 + j0 + 16);
}

__device__ __forceinline__ void gemm_step(const ABf16& pa, const ABf16& c0v,
                                          const ABf16& c1v, const ABf16& c2v,
                                          const ABf16& b0, const ABf16& b1,
                                          v8f (&acc)[6]) {
  ABf16 a0, a1, a2;
#pragma unroll
  for (int d = 0; d < 8; ++d) {
    a0.d[d] = pk_mul_bf16(c0v.d[d], pa.d[d]);
    a1.d[d] = pk_mul_bf16(c1v.d[d], pa.d[d]);
    a2.d[d] = pk_mul_bf16(c2v.d[d], pa.d[d]);
  }
  acc[0] = __builtin_amdgcn_wmma_f32_16x16x32_bf16(false, a0.v, false, b0.v, (short)0, acc[0], false, false);
  acc[1] = __builtin_amdgcn_wmma_f32_16x16x32_bf16(false, a1.v, false, b0.v, (short)0, acc[1], false, false);
  acc[2] = __builtin_amdgcn_wmma_f32_16x16x32_bf16(false, a2.v, false, b0.v, (short)0, acc[2], false, false);
  acc[3] = __builtin_amdgcn_wmma_f32_16x16x32_bf16(false, a0.v, false, b1.v, (short)0, acc[3], false, false);
  acc[4] = __builtin_amdgcn_wmma_f32_16x16x32_bf16(false, a1.v, false, b1.v, (short)0, acc[4], false, false);
  acc[5] = __builtin_amdgcn_wmma_f32_16x16x32_bf16(false, a2.v, false, b1.v, (short)0, acc[5], false, false);
}

__global__ void __launch_bounds__(32, 1)
k_cheb_gemm(const bf16_t* __restrict__ P, const bf16_t* __restrict__ chebB,
            const bf16_t* __restrict__ xtaT, float* __restrict__ rhs) {
  const int i0  = blockIdx.x * 16;
  const int c0  = blockIdx.y * 32;        /* two c-tiles per wave */
  const int b   = blockIdx.z;
  const int lane = threadIdx.x & 31;
  const int col  = lane & 15;
  const int hi   = lane >> 4;
  const int ka   = hi * 8;
  const int kb   = hi * 8;

  const bf16_t* prow  = P    + ((size_t)b * N_ + (i0 + col)) * N_ + ka;
  const bf16_t* brow0 = xtaT + ((size_t)b * C_ + (c0 + col)) * N_ + kb;
  const bf16_t* brow1 = brow0 + (size_t)16 * N_;
  const bf16_t* crow0 = chebB + (size_t)(i0 + col) * N_ + ka;
  const bf16_t* crow1 = crow0 + (size_t)N_ * N_;
  const bf16_t* crow2 = crow1 + (size_t)N_ * N_;

  v8f acc[6] = {};
  ABf16 paA, c0A, c1A, c2A, b0A, b1A;
  ABf16 paB, c0B, c1B, c2B, b0B, b1B;
  gemm_load(prow, crow0, crow1, crow2, brow0, brow1, 0, paA, c0A, c1A, c2A, b0A, b1A);
  for (int j0 = 0; j0 < N_; j0 += 64) {
    gemm_load(prow, crow0, crow1, crow2, brow0, brow1, j0 + 32,
              paB, c0B, c1B, c2B, b0B, b1B);
    gemm_step(paA, c0A, c1A, c2A, b0A, b1A, acc);
    if (j0 + 64 < N_)
      gemm_load(prow, crow0, crow1, crow2, brow0, brow1, j0 + 64,
                paA, c0A, c1A, c2A, b0A, b1A);
    gemm_step(paB, c0B, c1B, c2B, b0B, b1B, acc);
  }

  /* store: C-layout lane holds rows i0+ka..i0+ka+7, col -> contiguous i.
     Non-temporal: rhs is stream-once, keep cheb/P resident in L2. */
  const size_t kstride = (size_t)B_ * C_ * N_;
#pragma unroll
  for (int ct = 0; ct < 2; ++ct) {
    size_t base = ((size_t)b * C_ + (c0 + ct * 16 + col)) * N_ + i0 + ka;
#pragma unroll
    for (int k = 0; k < KCH; ++k) {
      Acc8 u;
      u.v = acc[ct * 3 + k];
      __builtin_nontemporal_store(u.f4[0], (f32x4*)(rhs + k * kstride + base));
      __builtin_nontemporal_store(u.f4[1], (f32x4*)(rhs + k * kstride + base + 4));
    }
  }
}

/* ---------- theta projection + relu: xgc bf16 [b][t][n][o] ---------- */
__global__ void k_theta(const float* __restrict__ rhs, const float* __restrict__ theta,
                        bf16_t* __restrict__ xgc) {
  __shared__ float rhsL[KCH * FIN * 32];   /* [k][f][nl] */
  __shared__ float thL[KCH * FIN * FOUT];  /* [k][f][o]  */
  const int n0 = blockIdx.x * 32;
  const int t  = blockIdx.y;
  const int b  = blockIdx.z;
  const int tid = threadIdx.x;
  for (int e = tid; e < KCH * FIN * 32; e += 256) {
    int k = e >> 10, r = e & 1023;
    int f = r >> 5, nl = r & 31;
    rhsL[e] = __builtin_nontemporal_load(
        rhs + ((size_t)(k * B_ + b) * C_ + t * FIN + f) * N_ + n0 + nl);
  }
  for (int e = tid; e < KCH * FIN * FOUT; e += 256) thL[e] = theta[e];
  __syncthreads();
  const int o  = tid & 63;
  const int nb = tid >> 6;
#pragma unroll
  for (int m = 0; m < 8; ++m) {
    int nl = nb * 8 + m;
    float acc = 0.f;
#pragma unroll
    for (int kf = 0; kf < KCH * FIN; ++kf)
      acc += thL[kf * FOUT + o] * rhsL[kf * 32 + nl];
    xgc[(((size_t)b * T_ + t) * N_ + n0 + nl) * FOUT + o] = (bf16_t)fmaxf(acc, 0.f);
  }
}

/* ----- fused temporal conv + residual + relu + layernorm (WMMA) ----- */
__global__ void __launch_bounds__(32, 1)
k_final(const bf16_t* __restrict__ xgc, const bf16_t* __restrict__ xb,
        const bf16_t* __restrict__ Wt, const bf16_t* __restrict__ rwT,
        const float* __restrict__ tconv_b, const float* __restrict__ rconv_b,
        const float* __restrict__ ln_g, const float* __restrict__ ln_b,
        float* __restrict__ out) {
  const int i0 = blockIdx.x * 16;
  const int t  = blockIdx.y;
  const int b  = blockIdx.z;
  const int lane = threadIdx.x & 31;
  const int col  = lane & 15;
  const int hi   = lane >> 4;
  const int ka   = hi * 8;
  const int kb   = hi * 8;

  v8f acc[4] = {};
#pragma unroll
  for (int s = 0; s < 6; ++s) {
    const int dt  = s >> 1;
    const int ci0 = (s & 1) * 32;
    const int tt  = t + dt - 1;
    ABf16 a;
    if (tt >= 0 && tt < T_) {
      const bf16_t* ap = xgc + (((size_t)b * T_ + tt) * N_ + (i0 + col)) * FOUT + ci0 + ka;
      a.q[0] = *(const uint4*)ap;
      a.q[1] = *(const uint4*)(ap + 16);
    } else {
#pragma unroll
      for (int d = 0; d < 8; ++d) a.d[d] = 0u;
    }
#pragma unroll
    for (int c = 0; c < 4; ++c) {
      ABf16 w;
      const bf16_t* wp = Wt + (size_t)(c * 16 + col) * 192 + s * 32 + kb;
      w.q[0] = *(const uint4*)wp;
      w.q[1] = *(const uint4*)(wp + 16);
      acc[c] = __builtin_amdgcn_wmma_f32_16x16x32_bf16(false, a.v, false, w.v, (short)0, acc[c], false, false);
    }
  }
  { /* residual 1x1 conv: x @ rconv_w^T, K = 32 */
    ABf16 a;
    const bf16_t* ap = xb + (((size_t)b * T_ + t) * N_ + (i0 + col)) * FIN + ka;
    a.q[0] = *(const uint4*)ap;
    a.q[1] = *(const uint4*)(ap + 16);
#pragma unroll
    for (int c = 0; c < 4; ++c) {
      ABf16 w;
      const bf16_t* wp = rwT + (size_t)(c * 16 + col) * FIN + kb;
      w.q[0] = *(const uint4*)wp;
      w.q[1] = *(const uint4*)(wp + 16);
      acc[c] = __builtin_amdgcn_wmma_f32_16x16x32_bf16(false, a.v, false, w.v, (short)0, acc[c], false, false);
    }
  }
  float bias[4], gg[4], bb[4];
#pragma unroll
  for (int c = 0; c < 4; ++c) {
    int o = c * 16 + col;
    bias[c] = tconv_b[o] + rconv_b[o];
    gg[c] = ln_g[o];
    bb[c] = ln_b[o];
  }
#pragma unroll
  for (int v = 0; v < 8; ++v) {
    float r0 = fmaxf(acc[0][v] + bias[0], 0.f);
    float r1 = fmaxf(acc[1][v] + bias[1], 0.f);
    float r2 = fmaxf(acc[2][v] + bias[2], 0.f);
    float r3 = fmaxf(acc[3][v] + bias[3], 0.f);
    float s1 = r0 + r1 + r2 + r3;
    float s2 = r0 * r0 + r1 * r1 + r2 * r2 + r3 * r3;
#pragma unroll
    for (int m = 1; m < 16; m <<= 1) {      /* reduce within the 16-lane half */
      s1 += __shfl_xor(s1, m, 32);
      s2 += __shfl_xor(s2, m, 32);
    }
    float mu  = s1 * (1.f / 64.f);
    float var = s2 * (1.f / 64.f) - mu * mu;
    float rs  = rsqrtf(var + 1e-5f);
    size_t base = (((size_t)b * T_ + t) * N_ + (i0 + v + ka)) * FOUT + col;
    out[base]      = (r0 - mu) * rs * gg[0] + bb[0];
    out[base + 16] = (r1 - mu) * rs * gg[1] + bb[1];
    out[base + 32] = (r2 - mu) * rs * gg[2] + bb[2];
    out[base + 48] = (r3 - mu) * rs * gg[3] + bb[3];
  }
}

extern "C" void kernel_launch(void* const* d_in, const int* in_sizes, int n_in,
                              void* d_out, int out_size, void* d_ws, size_t ws_size,
                              hipStream_t stream) {
  const float* x       = (const float*)d_in[0];
  const float* cheb    = (const float*)d_in[1];
  const float* wq_t    = (const float*)d_in[2];
  const float* wk_t    = (const float*)d_in[3];
  const float* wq_s    = (const float*)d_in[4];
  const float* wk_s    = (const float*)d_in[5];
  const float* theta   = (const float*)d_in[6];
  const float* tconv_w = (const float*)d_in[7];
  const float* tconv_b = (const float*)d_in[8];
  const float* rconv_w = (const float*)d_in[9];
  const float* rconv_b = (const float*)d_in[10];
  const float* ln_g    = (const float*)d_in[11];
  const float* ln_b    = (const float*)d_in[12];
  float* out = (float*)d_out;

  char* ws = (char*)d_ws;
  size_t cur = 0;
  auto alloc = [&](size_t bytes) -> char* {
    char* p = ws + cur;
    cur = (cur + bytes + 255) & ~(size_t)255;
    return p;
  };
  float*  qt    = (float*)alloc((size_t)B_ * T_ * N_ * 4);
  float*  kt    = (float*)alloc((size_t)B_ * T_ * N_ * 4);
  float*  ta    = (float*)alloc((size_t)B_ * T_ * T_ * 4);
  float*  qsT   = (float*)alloc((size_t)B_ * N_ * T_ * 4);
  float*  ksT   = (float*)alloc((size_t)B_ * N_ * T_ * 4);
  bf16_t* xtaT  = (bf16_t*)alloc((size_t)B_ * C_ * N_ * 2);
  bf16_t* Pbuf  = (bf16_t*)alloc((size_t)B_ * N_ * N_ * 2);
  bf16_t* chebB = (bf16_t*)alloc((size_t)KCH * N_ * N_ * 2);
  bf16_t* xb    = (bf16_t*)alloc((size_t)B_ * T_ * N_ * FIN * 2);
  bf16_t* Wt    = (bf16_t*)alloc((size_t)FOUT * 192 * 2);
  bf16_t* rwT   = (bf16_t*)alloc((size_t)FOUT * FIN * 2);
  float*  rhs   = (float*)alloc((size_t)KCH * B_ * C_ * N_ * 4);
  bf16_t* xgc   = (bf16_t*)alloc((size_t)B_ * T_ * N_ * FOUT * 2);

  /* conversions */
  {
    int n = KCH * N_ * N_;
    k_cvt_bf16<<<(n + 255) / 256, 256, 0, stream>>>(cheb, chebB, n);
  }
  {
    int n = B_ * T_ * N_ * FIN;
    k_cvt_bf16<<<(n + 255) / 256, 256, 0, stream>>>(x, xb, n);
  }
  {
    int n = FOUT * FIN;
    k_cvt_bf16<<<(n + 255) / 256, 256, 0, stream>>>(rconv_w, rwT, n);
  }
  k_prep_wt<<<(FOUT * 192 + 255) / 256, 256, 0, stream>>>(tconv_w, Wt);

  /* temporal attention */
  k_qtkt<<<(B_ * T_ * N_) / 256, 256, 0, stream>>>(x, wq_t, wk_t, qt, kt);
  k_ta<<<B_, 256, 0, stream>>>(qt, kt, ta);
  k_xta<<<dim3(N_ / 8, B_), 256, 0, stream>>>(x, ta, wq_s, wk_s, xtaT, qsT, ksT);

  /* spatial attention softmax */
  k_sa<<<dim3(N_ / 8, B_), 256, 0, stream>>>(qsT, ksT, Pbuf);

  /* main WMMA GEMM: 128 i-tiles x 12 c-tile-pairs x 16 batches, 1 wave each */
  k_cheb_gemm<<<dim3(N_ / 16, C_ / 32, B_), 32, 0, stream>>>(Pbuf, chebB, xtaT, rhs);

  /* theta projection + relu */
  k_theta<<<dim3(N_ / 32, T_, B_), 256, 0, stream>>>(rhs, theta, xgc);

  /* fused temporal conv + residual + relu + layernorm */
  k_final<<<dim3(N_ / 16, T_, B_), 32, 0, stream>>>(xgc, xb, Wt, rwT, tconv_b, rconv_b,
                                                    ln_g, ln_b, out);
}
